// LSTM_77627238908736
// MI455X (gfx1250) — matrix operand order repaired and checked by reading, not compiled, not run on hardware
//
#include <hip/hip_runtime.h>
#include <hip/hip_bf16.h>

// ---------------------------------------------------------------------------
// LSTM (with source's c-overwrite bug) for MI455X / gfx1250, wave32 + WMMA.
//   Phase 1: xp[t,b,g,h] = x @ U_w^T + U_b + b      (bf16 WMMA, f32 accum)
//            WG tile 128x128, wave tile 32x64 (8 accumulators) -> 1.5 loads/WMMA
//   Phase 2: persistent recurrence kernel, 2048 steps, grid barrier per step,
//            V_w held entirely in VGPRs (16 x v16bf = 128 VGPRs per lane).
// ---------------------------------------------------------------------------

typedef __bf16 bf16_t;
typedef __attribute__((ext_vector_type(16))) __bf16 v16bf;
typedef __attribute__((ext_vector_type(8)))  __bf16 v8bf;
typedef __attribute__((ext_vector_type(8)))  float  v8f;

#define LSTM_B   32
#define LSTM_T   2048
#define LSTM_I   512
#define LSTM_H   512
#define NGATE    4
#define GCOLS    (NGATE * LSTM_H)          // 2048
#define MROWS    (LSTM_B * LSTM_T)         // 65536
#define NWG_REC  32                        // workgroups in recurrence kernel

// ---- workspace layout (bytes) ----
#define XP_OFF     0ull
#define XP_BYTES   ((size_t)LSTM_T * LSTM_B * GCOLS * 4)      // 536,870,912
#define XBF_OFF    (XP_OFF + XP_BYTES)
#define XBF_BYTES  ((size_t)LSTM_B * LSTM_T * LSTM_I * 2)     //  67,108,864
#define UW_OFF     (XBF_OFF + XBF_BYTES)
#define UW_BYTES   ((size_t)NGATE * LSTM_H * LSTM_I * 2)      //   2,097,152
#define VW_OFF     (UW_OFF + UW_BYTES)
#define VW_BYTES   ((size_t)NGATE * LSTM_H * LSTM_H * 2)      //   2,097,152
#define HBUF_OFF   (VW_OFF + VW_BYTES)
#define HBUF_BYTES ((size_t)LSTM_B * LSTM_H * 2)              //      32,768
#define CNT_OFF    (HBUF_OFF + HBUF_BYTES)                    // int counter

// ---------------------------------------------------------------------------
// Fragment loaders (per ISA 7.12.2 VGPR layouts, wave32).
// A 16x32 bf16: lanes 0-15 row M=lane, K = {k0..k0+7, k0+16..k0+23};
//               lanes 16-31 row M=lane-16, K = {k0+8..15, k0+24..31}.
// B 32x16 bf16: lanes 0-15 col N=lane,  K = {k0..k0+7, k0+16..k0+23};
//               lanes 16-31 col N=lane-16, K = {k0+8..15, k0+24..31}.
// Both are K-contiguous per row/col in our memory layouts -> 16B loads.
// ---------------------------------------------------------------------------
__device__ __forceinline__ v16bf load_a_frag(const bf16_t* rowp, int k0, int half) {
  v8bf lo = *(const v8bf*)(rowp + k0 + half * 8);
  v8bf hi = *(const v8bf*)(rowp + k0 + half * 8 + 16);
  return __builtin_shufflevector(lo, hi, 0,1,2,3,4,5,6,7,8,9,10,11,12,13,14,15);
}
__device__ __forceinline__ v16bf load_b_frag(const bf16_t* colp, int k0, int half) {
  v8bf lo = *(const v8bf*)(colp + k0 + half * 8);
  v8bf hi = *(const v8bf*)(colp + k0 + half * 8 + 16);
  return __builtin_shufflevector(lo, hi, 0,1,2,3,4,5,6,7,8,9,10,11,12,13,14,15);
}
__device__ __forceinline__ v8f wmma_bf16(v16bf a, v16bf b, v8f c) {
  return __builtin_amdgcn_wmma_f32_16x16x32_bf16(
      /*neg_a=*/false, a, /*neg_b=*/false, b,
      /*c_mod=*/(short)0, c, /*reuse_a=*/false, /*reuse_b=*/false);
}
__device__ __forceinline__ float sigmoidf_fast(float x) {
  return 1.0f / (1.0f + __expf(-x));
}

// ---------------------------------------------------------------------------
// f32 -> bf16 conversion (8 elements / thread)
// ---------------------------------------------------------------------------
__global__ void lstm_cvt_bf16(const float* __restrict__ in,
                              bf16_t* __restrict__ out, int n) {
  int i = (blockIdx.x * blockDim.x + threadIdx.x) * 8;
  if (i < n) {
#pragma unroll
    for (int j = 0; j < 8; ++j) out[i + j] = (bf16_t)in[i + j];
  }
}

// ---------------------------------------------------------------------------
// Phase 1: xp[(t*B + b)*4H + n] = sum_k x[b,t,k] * U_w[n,k] + U_b[n] + b[n]
// Workgroup = 8 waves, tile 128(M) x 128(N); wave = 32(M) x 64(N) = 8 C tiles.
// Per K-step: 4 A-loads + 8 B-loads (16B each) feed 8 WMMAs.
// ---------------------------------------------------------------------------
__global__ __launch_bounds__(256)
void lstm_gemm_in(const bf16_t* __restrict__ xbf, const bf16_t* __restrict__ Uw,
                  const float* __restrict__ Ub, const float* __restrict__ bb,
                  float* __restrict__ xp) {
  const int tid  = threadIdx.x;
  const int lane = tid & 31;
  const int w    = tid >> 5;
  const int rc   = lane & 15;
  const int half = lane >> 4;
  const int rowBase = blockIdx.y * 128 + (w & 3) * 32;   // 2 M-tiles per wave
  const int colBase = blockIdx.x * 128 + (w >> 2) * 64;  // 4 N-tiles per wave

  const bf16_t* arow0 = xbf + (size_t)(rowBase + rc) * LSTM_I;
  const bf16_t* arow1 = xbf + (size_t)(rowBase + 16 + rc) * LSTM_I;

  v8f acc[2][4];
#pragma unroll
  for (int mi = 0; mi < 2; ++mi)
#pragma unroll
    for (int i = 0; i < 4; ++i) acc[mi][i] = v8f{};

#pragma unroll 2
  for (int ks = 0; ks < LSTM_I / 32; ++ks) {
    const int k0 = ks * 32;
    v16bf a0 = load_a_frag(arow0, k0, half);
    v16bf a1 = load_a_frag(arow1, k0, half);
#pragma unroll
    for (int i = 0; i < 4; ++i) {
      const bf16_t* bcol = Uw + (size_t)(colBase + i * 16 + rc) * LSTM_I;
      v16bf b = load_b_frag(bcol, k0, half);
      acc[0][i] = wmma_bf16(a0, b, acc[0][i]);
      acc[1][i] = wmma_bf16(a1, b, acc[1][i]);
    }
  }

#pragma unroll
  for (int i = 0; i < 4; ++i) {
    const int col  = colBase + i * 16 + rc;
    const float bias = Ub[col] + bb[col];
#pragma unroll
    for (int mi = 0; mi < 2; ++mi) {
#pragma unroll
      for (int j = 0; j < 8; ++j) {
        const int row = rowBase + mi * 16 + half * 8 + j;  // row = b*T + t
        const int bb_ = row >> 11;                         // / T
        const int tt  = row & (LSTM_T - 1);                // % T
        xp[((size_t)tt * LSTM_B + bb_) * GCOLS + col] = acc[mi][i][j] + bias;
      }
    }
  }
}

// ---------------------------------------------------------------------------
// Phase 2: persistent recurrence. 32 WGs x 256 threads (8 waves).
// WG owns h-columns [hbase, hbase+16). Wave w -> (Mtile m = w>>2, gate g = w&3).
// B operand (V_w slice) preloaded into 16 v16bf registers per lane.
// ---------------------------------------------------------------------------
__global__ __launch_bounds__(256)
void lstm_recur(const float* __restrict__ xp, const bf16_t* __restrict__ Vw,
                const float* __restrict__ Vb, bf16_t* __restrict__ hbuf,
                float* __restrict__ out, int* __restrict__ bar_cnt) {
  __shared__ float ex[LSTM_B][68];   // pre-activation exchange, padded stride

  const int tid  = threadIdx.x;
  const int lane = tid & 31;
  const int w    = tid >> 5;
  const int m    = w >> 2;           // 0..1 : M tile (batch rows 16m..16m+15)
  const int g    = w & 3;            // gate
  const int rc   = lane & 15;
  const int half = lane >> 4;
  const int hbase = blockIdx.x * 16;

  // --- preload V_w tile into registers: column = (g, hbase+rc), all K=512 ---
  const bf16_t* bcol = Vw + (size_t)(g * LSTM_H + hbase + rc) * LSTM_H;
  v16bf bfrag[16];
#pragma unroll
  for (int ks = 0; ks < 16; ++ks) bfrag[ks] = load_b_frag(bcol, ks * 32, half);

  const float vb = Vb[g * LSTM_H + hbase + rc];
  const bf16_t* arow = hbuf + (size_t)(m * 16 + rc) * LSTM_H;

  for (int t = 0; t < LSTM_T; ++t) {
    // prefetch next step's xp slice for this wave (speculative)
    {
      const int tn = (t + 1 < LSTM_T) ? t + 1 : t;
      __builtin_prefetch(
          xp + (((size_t)tn * LSTM_B + m * 16) * NGATE + g) * LSTM_H + hbase, 0, 0);
    }

    // --- GEMM: acc = h @ V_w[g]^T for this 16x16 tile ---
    v8f acc = v8f{};
#pragma unroll
    for (int ks = 0; ks < 16; ++ks) {
      v16bf a = load_a_frag(arow, ks * 32, half);
      acc = wmma_bf16(a, bfrag[ks], acc);
    }

    // --- write pre-activations (+xp +V_b) to LDS exchange tile ---
#pragma unroll
    for (int j = 0; j < 8; ++j) {
      const int row = m * 16 + half * 8 + j;     // batch index 0..31
      const float pre = acc[j] + vb +
          xp[(((size_t)t * LSTM_B + row) * NGATE + g) * LSTM_H + hbase + rc];
      ex[row][g * 16 + rc] = pre;
    }
    __syncthreads();

    // --- fused gates: 512 elements / 256 threads ---
#pragma unroll
    for (int r = 0; r < 2; ++r) {
      const int e   = tid + r * 256;
      const int row = e >> 4;
      const int col = e & 15;
      const float gi = ex[row][ 0 + col];
      const float gf = ex[row][16 + col];
      const float gc = ex[row][32 + col];
      const float go = ex[row][48 + col];
      const float it   = sigmoidf_fast(gi);
      const float ft   = sigmoidf_fast(gf);
      const float cand = tanhf(gc);
      const float ot   = sigmoidf_fast(go);
      const float c = (ft + it) * cand;      // faithful to source bug
      const float h = ot * tanhf(c);

      out[(size_t)t * LSTM_B * LSTM_H + (size_t)row * LSTM_H + hbase + col] = h;
      hbuf[(size_t)row * LSTM_H + hbase + col] = (bf16_t)h;
      if (t == LSTM_T - 1) {
        const size_t base = (size_t)LSTM_T * LSTM_B * LSTM_H;
        out[base + (size_t)row * LSTM_H + hbase + col] = h;                       // h_t
        out[base + LSTM_B * LSTM_H + (size_t)row * LSTM_H + hbase + col] = c;     // c_t
      }
    }

    // --- grid barrier (monotonic counter; reset by memset each launch) ---
    __threadfence();
    __syncthreads();
    if (tid == 0) {
      __hip_atomic_fetch_add(bar_cnt, 1, __ATOMIC_RELEASE, __HIP_MEMORY_SCOPE_AGENT);
      const int target = NWG_REC * (t + 1);
      while (__hip_atomic_load(bar_cnt, __ATOMIC_ACQUIRE, __HIP_MEMORY_SCOPE_AGENT) < target)
        __builtin_amdgcn_s_sleep(2);
    }
    __syncthreads();
    __threadfence();
  }
}

// ---------------------------------------------------------------------------
extern "C" void kernel_launch(void* const* d_in, const int* in_sizes, int n_in,
                              void* d_out, int out_size, void* d_ws, size_t ws_size,
                              hipStream_t stream) {
  const float* x   = (const float*)d_in[0];   // [B,T,I]
  const float* U_w = (const float*)d_in[1];   // [4,H,I]
  const float* U_b = (const float*)d_in[2];   // [4,H]
  const float* V_w = (const float*)d_in[3];   // [4,H,H]
  const float* V_b = (const float*)d_in[4];   // [4,H]
  const float* bg  = (const float*)d_in[5];   // [4,H]
  float* out = (float*)d_out;

  char* ws = (char*)d_ws;
  float*  xp   = (float*) (ws + XP_OFF);
  bf16_t* xbf  = (bf16_t*)(ws + XBF_OFF);
  bf16_t* Uwbf = (bf16_t*)(ws + UW_OFF);
  bf16_t* Vwbf = (bf16_t*)(ws + VW_OFF);
  bf16_t* hbuf = (bf16_t*)(ws + HBUF_OFF);
  int*    cnt  = (int*)   (ws + CNT_OFF);

  // convert inputs to bf16
  {
    const int nx = LSTM_B * LSTM_T * LSTM_I;
    lstm_cvt_bf16<<<nx / (8 * 256), 256, 0, stream>>>(x, xbf, nx);
    const int nw = NGATE * LSTM_H * LSTM_I;
    lstm_cvt_bf16<<<nw / (8 * 256), 256, 0, stream>>>(U_w, Uwbf, nw);
    lstm_cvt_bf16<<<nw / (8 * 256), 256, 0, stream>>>(V_w, Vwbf, nw);
  }

  // zero h0 and the barrier counter (hbuf and counter are adjacent)
  hipMemsetAsync(ws + HBUF_OFF, 0, HBUF_BYTES + sizeof(int), stream);

  // phase 1: input projection GEMM  (grid 16 x 512, WG tile 128x128)
  {
    dim3 grid(GCOLS / 128, MROWS / 128);
    lstm_gemm_in<<<grid, 256, 0, stream>>>(xbf, Uwbf, U_b, bg, xp);
  }

  // phase 2: persistent recurrence
  lstm_recur<<<NWG_REC, 256, 0, stream>>>(xp, Vwbf, V_b, hbuf, out, cnt);
}
